// GATConvModule_70952859730404
// MI455X (gfx1250) — matrix operand order repaired and checked by reading, not compiled, not run on hardware
//
#include <hip/hip_runtime.h>
#include <math.h>

#define IN_F   256
#define OUT_F  128
#define HEADS  4
#define HC     (HEADS*OUT_F)   // 512
#define NEG_SLOPE 0.2f

typedef __attribute__((ext_vector_type(16))) _Float16 v16h;
typedef __attribute__((ext_vector_type(8)))  _Float16 v8h;
typedef __attribute__((ext_vector_type(8)))  float    v8f;

// ---------------- precision-convert kernels ----------------
__global__ void cvt_x_f16(const float* __restrict__ x, _Float16* __restrict__ xh, int n) {
  int i = blockIdx.x * blockDim.x + threadIdx.x;
  if (i < n) xh[i] = (_Float16)x[i];
}

// W [256][512] f32 -> Wt [512][256] f16 (transposed: B columns contiguous)
__global__ void cvt_w_f16_t(const float* __restrict__ W, _Float16* __restrict__ wt, int n) {
  int i = blockIdx.x * blockDim.x + threadIdx.x;
  if (i < n) {
    int c = i / IN_F, k = i % IN_F;
    wt[i] = (_Float16)W[(size_t)k * HC + c];
  }
}

// ---------------- WMMA GEMM: h[N,512] = x[N,256] @ W[256,512] ----------------
// One wave32 per 16x16 output tile; K-loop of 8 x v_wmma_f32_16x16x32_f16.
__global__ void gemm_wmma(const _Float16* __restrict__ xh, const _Float16* __restrict__ wt,
                          float* __restrict__ h, int nodes, int tiles_m) {
  int wave = threadIdx.x >> 5;
  int tile = blockIdx.x * 8 + wave;
  int tiles_total = tiles_m * (HC / 16);
  if (tile >= tiles_total) return;

  int tm = tile >> 5;        // / (HC/16)=32
  int tn = tile & 31;
  int lane = threadIdx.x & 31;
  int g = lane >> 4;         // lane group (0/1)
  int r = lane & 15;

  int row = tm * 16 + r;
  int rowc = row < nodes ? row : nodes - 1;
  const _Float16* arow = xh + (size_t)rowc * IN_F;
  const _Float16* bcol = wt + (size_t)(tn * 16 + r) * IN_F;

  v8f c = {};
  #pragma unroll
  for (int k0 = 0; k0 < IN_F; k0 += 32) {
    // A 16x32 f16 layout: lane holds row r, K = k0+g*16 .. +15 contiguous
    v8h a0 = *(const v8h*)(arow + k0 + g * 16);
    v8h a1 = *(const v8h*)(arow + k0 + g * 16 + 8);
    // B 32x16 f16 layout: lane holds column r, K = k0+g*16 .. +15 contiguous
    v8h b0 = *(const v8h*)(bcol + k0 + g * 16);
    v8h b1 = *(const v8h*)(bcol + k0 + g * 16 + 8);
    v16h A, B;
    #pragma unroll
    for (int i = 0; i < 8; ++i) { A[i] = a0[i]; A[8 + i] = a1[i]; B[i] = b0[i]; B[8 + i] = b1[i]; }
    c = __builtin_amdgcn_wmma_f32_16x16x32_f16(false, A, false, B, (short)0, c, false, false);
  }

  // D layout: lane -> column (tn*16+r), VGPR v -> row (tm*16 + g*8 + v)
  int col = tn * 16 + r;
  #pragma unroll
  for (int v = 0; v < 8; ++v) {
    int m = tm * 16 + g * 8 + v;
    if (m < nodes) h[(size_t)m * HC + col] = c[v];
  }
}

// ---------------- per-node attention dots: a_s/a_d [N,H] ----------------
__global__ void attn_dots(const float* __restrict__ h, const float* __restrict__ att_src,
                          const float* __restrict__ att_dst, float* __restrict__ a_s,
                          float* __restrict__ a_d, int nodes) {
  int wave = threadIdx.x >> 5;
  int n = blockIdx.x * 8 + wave;
  if (n >= nodes) return;
  int lane = threadIdx.x & 31;
  const float* hrow = h + (size_t)n * HC;
  for (int hd = 0; hd < HEADS; ++hd) {
    float ps = 0.f, pd = 0.f;
    for (int cidx = lane; cidx < OUT_F; cidx += 32) {
      float v = hrow[hd * OUT_F + cidx];
      ps += v * att_src[hd * OUT_F + cidx];
      pd += v * att_dst[hd * OUT_F + cidx];
    }
    for (int m = 16; m >= 1; m >>= 1) {
      ps += __shfl_xor(ps, m, 32);
      pd += __shfl_xor(pd, m, 32);
    }
    if (lane == 0) { a_s[n * HEADS + hd] = ps; a_d[n * HEADS + hd] = pd; }
  }
}

// ---------------- order-preserving f32 <-> u32 encoding for atomicMax ----------------
__device__ __forceinline__ unsigned enc_f32(float x) {
  unsigned u = __float_as_uint(x);
  return (u & 0x80000000u) ? ~u : (u | 0x80000000u);
}
__device__ __forceinline__ float dec_f32(unsigned u) {
  return __uint_as_float((u & 0x80000000u) ? (u ^ 0x80000000u) : ~u);
}

// ---------------- edge pass A: alpha = leaky_relu(a_s[s]+a_d[d]); segment max ----------------
__global__ void edge_alpha(const int* __restrict__ src, const int* __restrict__ dst,
                           const float* __restrict__ a_s, const float* __restrict__ a_d,
                           float* __restrict__ alpha, unsigned* __restrict__ amax,
                           int E, int Et) {
  int e = blockIdx.x * blockDim.x + threadIdx.x;
  if (e >= Et) return;
  int s = (e < E) ? src[e] : (e - E);   // self loops appended
  int d = (e < E) ? dst[e] : (e - E);
  #pragma unroll
  for (int hd = 0; hd < HEADS; ++hd) {
    float al = a_s[s * HEADS + hd] + a_d[d * HEADS + hd];
    al = al > 0.f ? al : NEG_SLOPE * al;
    alpha[(size_t)e * HEADS + hd] = al;
    atomicMax(&amax[d * HEADS + hd], enc_f32(al));
  }
}

// ---------------- edge pass B: ex = exp(alpha - amax[d]); segment sum ----------------
__global__ void edge_exp(const int* __restrict__ dst, const unsigned* __restrict__ amax,
                         float* __restrict__ alpha, float* __restrict__ denom,
                         int E, int Et) {
  int e = blockIdx.x * blockDim.x + threadIdx.x;
  if (e >= Et) return;
  int d = (e < E) ? dst[e] : (e - E);
  #pragma unroll
  for (int hd = 0; hd < HEADS; ++hd) {
    float ex = expf(alpha[(size_t)e * HEADS + hd] - dec_f32(amax[d * HEADS + hd]));
    alpha[(size_t)e * HEADS + hd] = ex;     // reuse buffer for ex
    atomicAdd(&denom[d * HEADS + hd], ex);
  }
}

// ---------------- edge pass C: acc[d,c] += 1/4 * sum_h w[e,h] * h[s,h,c] ----------------
// One wave32 per edge; 4 channels per lane (float4 gather, 4 f32 atomics).
__global__ void edge_scatter(const int* __restrict__ src, const int* __restrict__ dst,
                             const float* __restrict__ h, const float* __restrict__ ex,
                             const float* __restrict__ denom, float* __restrict__ acc,
                             int E, int Et) {
  int wave = threadIdx.x >> 5;
  int e = blockIdx.x * 8 + wave;
  if (e >= Et) return;
  int lane = threadIdx.x & 31;
  int s = (e < E) ? src[e] : (e - E);
  int d = (e < E) ? dst[e] : (e - E);

  float w[HEADS];
  #pragma unroll
  for (int hd = 0; hd < HEADS; ++hd)
    w[hd] = 0.25f * ex[(size_t)e * HEADS + hd] / denom[d * HEADS + hd];

  const float* hs = h + (size_t)s * HC;
  int c0 = lane * 4;
  float r0 = 0.f, r1 = 0.f, r2 = 0.f, r3 = 0.f;
  #pragma unroll
  for (int hd = 0; hd < HEADS; ++hd) {
    const float4 v = *(const float4*)(hs + hd * OUT_F + c0);
    r0 += w[hd] * v.x; r1 += w[hd] * v.y; r2 += w[hd] * v.z; r3 += w[hd] * v.w;
  }
  float* ap = acc + (size_t)d * OUT_F + c0;
  atomicAdd(ap + 0, r0);
  atomicAdd(ap + 1, r1);
  atomicAdd(ap + 2, r2);
  atomicAdd(ap + 3, r3);
}

// ---------------- finalize: out = tanh(acc + bias) ----------------
__global__ void finalize(const float* __restrict__ acc, const float* __restrict__ bias,
                         float* __restrict__ out, int total) {
  int i = blockIdx.x * blockDim.x + threadIdx.x;
  if (i < total) {
    int c = i & (OUT_F - 1);
    out[i] = tanhf(acc[i] + bias[c]);
  }
}

extern "C" void kernel_launch(void* const* d_in, const int* in_sizes, int n_in,
                              void* d_out, int out_size, void* d_ws, size_t ws_size,
                              hipStream_t stream) {
  const float* x       = (const float*)d_in[0];
  const int*   ei      = (const int*)d_in[1];
  const float* W       = (const float*)d_in[2];
  const float* att_src = (const float*)d_in[3];
  const float* att_dst = (const float*)d_in[4];
  const float* bias    = (const float*)d_in[5];
  float* out = (float*)d_out;

  int nodes = in_sizes[0] / IN_F;
  int E     = in_sizes[1] / 2;
  int Et    = E + nodes;                 // with self loops
  const int* src = ei;
  const int* dst = ei + E;

  // workspace carve-up (256B aligned)
  char* ws = (char*)d_ws;
  size_t off = 0;
  auto carve = [&](size_t bytes) -> char* {
    char* p = ws + off;
    off = (off + bytes + 255) & ~(size_t)255;
    return p;
  };
  _Float16* xh    = (_Float16*)carve((size_t)nodes * IN_F * 2);
  _Float16* wt    = (_Float16*)carve((size_t)IN_F * HC * 2);
  float*    h     = (float*)   carve((size_t)nodes * HC * 4);
  float*    a_s   = (float*)   carve((size_t)nodes * HEADS * 4);
  float*    a_d   = (float*)   carve((size_t)nodes * HEADS * 4);
  float*    alpha = (float*)   carve((size_t)Et * HEADS * 4);
  unsigned* amax  = (unsigned*)carve((size_t)nodes * HEADS * 4);
  float*    denom = (float*)   carve((size_t)nodes * HEADS * 4);
  float*    acc   = (float*)   carve((size_t)nodes * OUT_F * 4);

  // zero-init accumulators each call (ws is poisoned once, never re-poisoned)
  hipMemsetAsync(amax,  0, (size_t)nodes * HEADS * 4, stream);  // 0 == encoded(-NaN) <= everything
  hipMemsetAsync(denom, 0, (size_t)nodes * HEADS * 4, stream);
  hipMemsetAsync(acc,   0, (size_t)nodes * OUT_F * 4, stream);

  int nx = nodes * IN_F;
  cvt_x_f16<<<(nx + 255) / 256, 256, 0, stream>>>(x, xh, nx);
  int nw = IN_F * HC;
  cvt_w_f16_t<<<(nw + 255) / 256, 256, 0, stream>>>(W, wt, nw);

  int tiles_m = (nodes + 15) / 16;
  int tiles   = tiles_m * (HC / 16);
  gemm_wmma<<<(tiles + 7) / 8, 256, 0, stream>>>(xh, wt, h, nodes, tiles_m);

  attn_dots<<<(nodes + 7) / 8, 256, 0, stream>>>(h, att_src, att_dst, a_s, a_d, nodes);

  edge_alpha<<<(Et + 255) / 256, 256, 0, stream>>>(src, dst, a_s, a_d, alpha, amax, E, Et);
  edge_exp<<<(Et + 255) / 256, 256, 0, stream>>>(dst, amax, alpha, denom, E, Et);
  edge_scatter<<<(Et + 7) / 8, 256, 0, stream>>>(src, dst, h, alpha, denom, acc, E, Et);

  int total = nodes * OUT_F;
  finalize<<<(total + 255) / 256, 256, 0, stream>>>(acc, bias, out, total);
}
